// AutoEncoder_31233002176556
// MI455X (gfx1250) — matrix operand order repaired and checked by reading, not compiled, not run on hardware
//
#include <hip/hip_runtime.h>
#include <hip/hip_bf16.h>

typedef __bf16 bf16_t;
typedef __attribute__((ext_vector_type(16))) __bf16 v16bf;
typedef __attribute__((ext_vector_type(8)))  __bf16 v8bf;
typedef __attribute__((ext_vector_type(8)))  float  v8f;

#define NEG_SLOPE 0.2f
#define SM_EPS    1e-16f

// ---------------------------------------------------------------------------
// Utility kernels
// ---------------------------------------------------------------------------

__global__ void fill_f32(float* __restrict__ p, float v, int n) {
    int i = blockIdx.x * blockDim.x + threadIdx.x;
    if (i < n) p[i] = v;
}

__global__ void cast_f32_bf16(const float* __restrict__ in, bf16_t* __restrict__ out, int n) {
    int i = blockIdx.x * blockDim.x + threadIdx.x;
    if (i < n) out[i] = (bf16_t)in[i];
}

// out[c, r] = bf16(in[r, c])   — stage weights transposed so GEMM B is K-contiguous
__global__ void cast_t_f32_bf16(const float* __restrict__ in, bf16_t* __restrict__ out,
                                int R, int Ccols) {
    int i = blockIdx.x * blockDim.x + threadIdx.x;
    if (i >= R * Ccols) return;
    int r = i / Ccols, c = i % Ccols;
    out[(size_t)c * R + r] = (bf16_t)in[i];
}

// alpha_s[i] = h[i,:] . a_src ; alpha_d[i] = h[i,:] . a_dst   (one thread/node)
__global__ void node_dot2(const float* __restrict__ h,
                          const float* __restrict__ a_s, const float* __restrict__ a_d,
                          float* __restrict__ as_out, float* __restrict__ ad_out,
                          int N, int F) {
    int i = blockIdx.x * blockDim.x + threadIdx.x;
    if (i >= N) return;
    const float* hr = h + (size_t)i * F;
    float s0 = 0.f, s1 = 0.f;
    for (int f = 0; f < F; ++f) {
        float hv = hr[f];
        s0 += hv * a_s[f];
        s1 += hv * a_d[f];
    }
    as_out[i] = s0;
    ad_out[i] = s1;
}

__device__ inline void atomicMaxF32(float* addr, float val) {
    unsigned int* ua = (unsigned int*)addr;
    unsigned int old = *ua;
    while (true) {
        float cur = __uint_as_float(old);
        if (cur >= val) break;
        unsigned int assumed = old;
        old = atomicCAS(ua, assumed, __float_as_uint(val));
        if (old == assumed) break;
    }
}

// e = leakyrelu(as[src]+ad[dst]); segment max into mbuf[dst]
__global__ void edge_max(const int* __restrict__ src, const int* __restrict__ dst,
                         int E, int N,
                         const float* __restrict__ as, const float* __restrict__ ad,
                         float* __restrict__ earr, float* __restrict__ mbuf) {
    int i = blockIdx.x * blockDim.x + threadIdx.x;
    int Etot = E + N;
    if (i >= Etot) return;
    int s = (i < E) ? src[i] : (i - E);
    int d = (i < E) ? dst[i] : (i - E);
    if ((unsigned)s >= (unsigned)N || (unsigned)d >= (unsigned)N) { earr[i] = 0.f; return; }
    float e = as[s] + ad[d];
    e = (e > 0.f) ? e : NEG_SLOPE * e;
    earr[i] = e;
    atomicMaxF32(mbuf + d, e);
}

// ex = exp(e - m[dst]); denom[dst] += ex
__global__ void edge_exp(const int* __restrict__ dst, int E, int N,
                         float* __restrict__ earr,
                         const float* __restrict__ mbuf, float* __restrict__ denom) {
    int i = blockIdx.x * blockDim.x + threadIdx.x;
    int Etot = E + N;
    if (i >= Etot) return;
    int d = (i < E) ? dst[i] : (i - E);
    if ((unsigned)d >= (unsigned)N) { earr[i] = 0.f; return; }
    float ex = expf(earr[i] - mbuf[d]);
    earr[i] = ex;
    atomicAdd(denom + d, ex);
}

// agg[dst,:] += (ex/(denom[dst]+eps)) * h[src,:]   (one block per edge)
__global__ void edge_agg(const int* __restrict__ src, const int* __restrict__ dst,
                         int E, int N,
                         const float* __restrict__ exarr, const float* __restrict__ denom,
                         const float* __restrict__ h, float* __restrict__ agg, int F) {
    int i = blockIdx.x;
    int s = (i < E) ? src[i] : (i - E);
    int d = (i < E) ? dst[i] : (i - E);
    if ((unsigned)s >= (unsigned)N || (unsigned)d >= (unsigned)N) return;
    float att = exarr[i] / (denom[d] + SM_EPS);
    const float* hs = h + (size_t)s * F;
    float* ag = agg + (size_t)d * F;
    for (int f = threadIdx.x; f < F; f += blockDim.x)
        atomicAdd(ag + f, att * hs[f]);
}

// out_f32 = agg + bias ; out_bf = bf16(out_f32)
__global__ void bias_cast(const float* __restrict__ agg, const float* __restrict__ bias,
                          float* __restrict__ outf, bf16_t* __restrict__ outb,
                          int total, int F) {
    int i = blockIdx.x * blockDim.x + threadIdx.x;
    if (i >= total) return;
    float v = agg[i] + bias[i % F];
    outf[i] = v;
    outb[i] = (bf16_t)v;
}

// ---------------------------------------------------------------------------
// Tiled WMMA bf16 NT-GEMM:  C[Md,Nd] = A[Md,Kd] * B[Nd,Kd]^T
// Both operands row-major with K innermost -> all fragment loads are b128.
//
// Fragment layouts per CDNA5 ISA 7.12.2 (wave32), hi = lane>>4, l15 = lane&15:
//   A (16x32 bf16): row M = l15; elements j map to K = 8*hi + (j<8 ? j : j+8)
//       -> two contiguous 8-elem (16B) chunks at K = 8*hi and K = 8*hi+16
//   B^T (32x16):    col N = l15; element j -> K = 8*hi + (j<8 ? j : j+8)
//       -> two contiguous 8-elem chunks at K = 8*hi and K = 8*hi+16
//   C (16x16 f32):  vgpr r -> row M = r + 8*hi, col N = l15
//
// Block = 128 threads (4 waves) -> 64x64 C block; each wave 32x32 as 2x2 tiles.
// ---------------------------------------------------------------------------

#define CAT16(lo, hi) __builtin_shufflevector(lo, hi, 0,1,2,3,4,5,6,7,8,9,10,11,12,13,14,15)

__global__ __launch_bounds__(128)
void gemm_nt_tiled(const bf16_t* __restrict__ A, const bf16_t* __restrict__ B,
                   float* __restrict__ C, int Md, int Nd, int Kd) {
    const int wid  = threadIdx.x >> 5;
    const int lane = threadIdx.x & 31;
    const int hi   = lane >> 4;
    const int l15  = lane & 15;
    const int rowBase = blockIdx.y * 64 + (wid >> 1) * 32;
    const int colBase = blockIdx.x * 64 + (wid &  1) * 32;

    const bf16_t* Ar0 = A + (size_t)(rowBase + l15) * Kd;
    const bf16_t* Ar1 = Ar0 + (size_t)16 * Kd;
    const bf16_t* Br0 = B + (size_t)(colBase + l15) * Kd;
    const bf16_t* Br1 = Br0 + (size_t)16 * Kd;

    v8f acc00 = {}, acc01 = {}, acc10 = {}, acc11 = {};

    for (int k0 = 0; k0 < Kd; k0 += 32) {
        const int ka = k0 + (hi << 3);    // A chunks: [ka, ka+8) and [ka+16, ka+24)
        const int kb = k0 + (hi << 3);    // B chunks: [kb, kb+8) and [kb+16, kb+24)

        v8bf a0lo = *(const v8bf*)(Ar0 + ka);
        v8bf a0hi = *(const v8bf*)(Ar0 + ka + 16);
        v8bf a1lo = *(const v8bf*)(Ar1 + ka);
        v8bf a1hi = *(const v8bf*)(Ar1 + ka + 16);
        v8bf b0lo = *(const v8bf*)(Br0 + kb);
        v8bf b0hi = *(const v8bf*)(Br0 + kb + 16);
        v8bf b1lo = *(const v8bf*)(Br1 + kb);
        v8bf b1hi = *(const v8bf*)(Br1 + kb + 16);

        if (k0 + 32 < Kd) {               // next K-slab into cache (global_prefetch)
            __builtin_prefetch(Ar0 + ka + 32, 0, 1);
            __builtin_prefetch(Ar1 + ka + 32, 0, 1);
            __builtin_prefetch(Br0 + kb + 32, 0, 1);
            __builtin_prefetch(Br1 + kb + 32, 0, 1);
        }

        v16bf a0 = CAT16(a0lo, a0hi);
        v16bf a1 = CAT16(a1lo, a1hi);
        v16bf b0 = CAT16(b0lo, b0hi);
        v16bf b1 = CAT16(b1lo, b1hi);

        acc00 = __builtin_amdgcn_wmma_f32_16x16x32_bf16(false, a0, false, b0,
                                                        (short)0, acc00, false, false);
        acc01 = __builtin_amdgcn_wmma_f32_16x16x32_bf16(false, a0, false, b1,
                                                        (short)0, acc01, false, false);
        acc10 = __builtin_amdgcn_wmma_f32_16x16x32_bf16(false, a1, false, b0,
                                                        (short)0, acc10, false, false);
        acc11 = __builtin_amdgcn_wmma_f32_16x16x32_bf16(false, a1, false, b1,
                                                        (short)0, acc11, false, false);
    }

    float* C0 = C + (size_t)(rowBase + hi * 8) * Nd + colBase + l15;
    float* C1 = C0 + (size_t)16 * Nd;
#pragma unroll
    for (int r = 0; r < 8; ++r) {
        C0[(size_t)r * Nd]      = acc00[r];
        C0[(size_t)r * Nd + 16] = acc01[r];
        C1[(size_t)r * Nd]      = acc10[r];
        C1[(size_t)r * Nd + 16] = acc11[r];
    }
}

// ---------------------------------------------------------------------------
// Host side
// ---------------------------------------------------------------------------

static inline int ceil_div(int a, int b) { return (a + b - 1) / b; }

// One GATConv layer.  Wt_bf is the transposed weight [F, K] (K contiguous).
static void gat_layer(const bf16_t* Ain_bf, int K, const bf16_t* Wt_bf, int F,
                      const float* a_srcv, const float* a_dstv, const float* biasv,
                      const int* srcIdx, const int* dstIdx, int E, int N,
                      float* hlin, float* agg, float* alpha_s, float* alpha_d,
                      float* mbuf, float* denom, float* earr,
                      float* out_f32, bf16_t* out_bf, hipStream_t stream) {
    const int Etot = E + N;
    // 1. h = A @ W = A @ Wt^T  (WMMA, all-b128 fragment loads)
    gemm_nt_tiled<<<dim3(F / 64, N / 64), 128, 0, stream>>>(Ain_bf, Wt_bf, hlin, N, F, K);
    // 2. attention logits per node
    node_dot2<<<ceil_div(N, 256), 256, 0, stream>>>(hlin, a_srcv, a_dstv,
                                                    alpha_s, alpha_d, N, F);
    // 3. init reductions
    fill_f32<<<ceil_div(N, 256), 256, 0, stream>>>(mbuf, -3.0e38f, N);
    fill_f32<<<ceil_div(N, 256), 256, 0, stream>>>(denom, 0.f, N);
    fill_f32<<<ceil_div(N * F, 256), 256, 0, stream>>>(agg, 0.f, N * F);
    // 4-6. segment softmax + weighted aggregation (atomics resolve in L2)
    edge_max<<<ceil_div(Etot, 256), 256, 0, stream>>>(srcIdx, dstIdx, E, N,
                                                      alpha_s, alpha_d, earr, mbuf);
    edge_exp<<<ceil_div(Etot, 256), 256, 0, stream>>>(dstIdx, E, N, earr, mbuf, denom);
    edge_agg<<<Etot, 128, 0, stream>>>(srcIdx, dstIdx, E, N, earr, denom, hlin, agg, F);
    // 7. bias + bf16 recast for next layer
    bias_cast<<<ceil_div(N * F, 256), 256, 0, stream>>>(agg, biasv, out_f32, out_bf,
                                                        N * F, F);
}

extern "C" void kernel_launch(void* const* d_in, const int* in_sizes, int n_in,
                              void* d_out, int out_size, void* d_ws, size_t ws_size,
                              hipStream_t stream) {
    const float* x          = (const float*)d_in[0];
    const int*   edge_index = (const int*)  d_in[1];
    const float* W_enc      = (const float*)d_in[2];
    const float* a_src_enc  = (const float*)d_in[3];
    const float* a_dst_enc  = (const float*)d_in[4];
    const float* b_enc      = (const float*)d_in[5];
    const float* W_attr     = (const float*)d_in[6];
    const float* a_src_attr = (const float*)d_in[7];
    const float* a_dst_attr = (const float*)d_in[8];
    const float* b_attr     = (const float*)d_in[9];
    const float* W_struct   = (const float*)d_in[10];
    const float* a_src_str  = (const float*)d_in[11];
    const float* a_dst_str  = (const float*)d_in[12];
    const float* b_struct   = (const float*)d_in[13];

    const int IN_DIM = 256, HID = 128;
    const int N = in_sizes[0] / IN_DIM;
    const int E = in_sizes[1] / 2;
    const int* srcIdx = edge_index;
    const int* dstIdx = edge_index + E;
    const int Etot = E + N;

    // ---- carve workspace ----
    char* wp = (char*)d_ws;
    auto carve = [&](size_t bytes) {
        char* p = wp;
        wp += (bytes + 255) & ~(size_t)255;
        return (void*)p;
    };
    float*  hlin     = (float*) carve((size_t)N * IN_DIM * 4);   // GEMM output (max F)
    float*  agg      = (float*) carve((size_t)N * IN_DIM * 4);   // aggregation (max F)
    float*  h_struct = (float*) carve((size_t)N * IN_DIM * 4);   // struct-layer f32 out
    bf16_t* x_bf     = (bf16_t*)carve((size_t)N * IN_DIM * 2);
    bf16_t* h_bf     = (bf16_t*)carve((size_t)N * HID    * 2);
    bf16_t* hs_bf    = (bf16_t*)carve((size_t)N * IN_DIM * 2);
    bf16_t* WencT_bf = (bf16_t*)carve((size_t)IN_DIM * HID * 2); // [HID, IN_DIM]
    bf16_t* WattrT_bf= (bf16_t*)carve((size_t)HID * IN_DIM * 2); // [IN_DIM, HID]
    bf16_t* WstrT_bf = (bf16_t*)carve((size_t)HID * IN_DIM * 2); // [IN_DIM, HID]
    float*  alpha_s  = (float*) carve((size_t)N * 4);
    float*  alpha_d  = (float*) carve((size_t)N * 4);
    float*  mbuf     = (float*) carve((size_t)N * 4);
    float*  denom    = (float*) carve((size_t)N * 4);
    float*  earr     = (float*) carve((size_t)Etot * 4);
    float*  h_f32    = (float*) carve((size_t)N * HID * 4);      // enc-layer f32 out

    float* x_out = (float*)d_out;                  // [N, 256]
    float* s_out = x_out + (size_t)N * IN_DIM;     // [N, N]

    // ---- stage bf16 activations + transposed bf16 weights ----
    cast_f32_bf16<<<ceil_div(N * IN_DIM, 256), 256, 0, stream>>>(x, x_bf, N * IN_DIM);
    cast_t_f32_bf16<<<ceil_div(IN_DIM * HID, 256), 256, 0, stream>>>(W_enc, WencT_bf,
                                                                     IN_DIM, HID);
    cast_t_f32_bf16<<<ceil_div(HID * IN_DIM, 256), 256, 0, stream>>>(W_attr, WattrT_bf,
                                                                     HID, IN_DIM);
    cast_t_f32_bf16<<<ceil_div(HID * IN_DIM, 256), 256, 0, stream>>>(W_struct, WstrT_bf,
                                                                     HID, IN_DIM);

    // ---- layer 1: encoder  (x[N,256] -> h[N,128]) ----
    gat_layer(x_bf, IN_DIM, WencT_bf, HID,
              a_src_enc, a_dst_enc, b_enc,
              srcIdx, dstIdx, E, N,
              hlin, agg, alpha_s, alpha_d, mbuf, denom, earr,
              h_f32, h_bf, stream);

    // ---- layer 2: attr decoder  (h[N,128] -> x_[N,256], straight into d_out) ----
    gat_layer(h_bf, HID, WattrT_bf, IN_DIM,
              a_src_attr, a_dst_attr, b_attr,
              srcIdx, dstIdx, E, N,
              hlin, agg, alpha_s, alpha_d, mbuf, denom, earr,
              x_out, /*bf16 sink (unused downstream)*/ x_bf, stream);

    // ---- layer 3: struct decoder  (h[N,128] -> h_[N,256]) ----
    gat_layer(h_bf, HID, WstrT_bf, IN_DIM,
              a_src_str, a_dst_str, b_struct,
              srcIdx, dstIdx, E, N,
              hlin, agg, alpha_s, alpha_d, mbuf, denom, earr,
              h_struct, hs_bf, stream);

    // ---- s_ = h_ @ h_^T  [N,N]  (dominant: 137 GFLOP on the WMMA pipe) ----
    gemm_nt_tiled<<<dim3(N / 64, N / 64), 128, 0, stream>>>(hs_bf, hs_bf, s_out,
                                                            N, N, IN_DIM);
}
